// BaselineKNNModel_58067957842008
// MI455X (gfx1250) — matrix operand order repaired and checked by reading, not compiled, not run on hardware
//
#include <hip/hip_runtime.h>
#include <hip/hip_bf16.h>
#include <stdint.h>

// ---------------- problem constants ----------------
#define NQ      2048      // queries
#define DIM     512       // feature dim
#define NE      100000    // embeddings
#define CHUNKS  25        // N chunks for phase 1
#define NPC     4000      // embeddings per chunk (25*4000 = 100000)
#define TILESPC 250       // NPC/16
#define WAVES   8         // waves per phase-1 workgroup (256 threads, 2 waves/SIMD)
#define MB      128       // queries per phase-1 workgroup
#define MTILES  16        // NQ/MB
#define LDSA_STRIDE 520   // bf16 elems per LDS A row (512 + 8 pad -> conflict-free b128)
#define SIM_STRIDE  17    // f32 elems per LDS sim row (16 + 1 pad)
#define MERGE_STRIDE 81   // f32/int elems per merge row (80 + 1 pad, gcd(81,64)=1)
#define CANDQ   (CHUNKS * 10)   // candidates per query entering phase 2 (250)

#define LDSA_BYTES  (MB * LDSA_STRIDE * 2)                 // 133120
#define SIM_BYTES   (WAVES * MB * SIM_STRIDE * 4)          // 69632
#define LDS_BYTES   (LDSA_BYTES + SIM_BYTES)               // 202752 <= 320KB WGP LDS
// merge overlay (reuses A region after a barrier): 2 * 128*81*4 = 82944 <= LDSA_BYTES

typedef __attribute__((ext_vector_type(16))) __bf16 v16bf;
typedef __attribute__((ext_vector_type(8)))  float  v8f;
typedef __attribute__((ext_vector_type(4)))  unsigned int u32x4;
typedef __attribute__((ext_vector_type(8)))  int          i32x8;
typedef __attribute__((ext_vector_type(4)))  int          i32x4;

// Tensor Data Mover staging of the A tile (clang-23 / therock 6-arg builtin form).
#if defined(__has_builtin)
#if __has_builtin(__builtin_amdgcn_tensor_load_to_lds) && \
    __has_builtin(__builtin_amdgcn_s_wait_tensorcnt)
#define USE_TDM 1
#endif
#endif
#ifndef USE_TDM
#define USE_TDM 0
#endif

union BF16x16 { v16bf v; uint32_t u[8]; uint4 q[2]; };

__device__ __forceinline__ uint32_t f2bf_bits(float f) {
  uint32_t u = __float_as_uint(f);
  return (u + 0x7FFFu + ((u >> 16) & 1u)) >> 16;   // RNE f32 -> bf16
}

// candidate "better" under jax top_k tie rule: larger value, then smaller index
#define BETTER(V, I, CV, CI) ((V) > (CV) || ((V) == (CV) && (I) < (CI)))

// insert (V,ID) into sorted-descending 10-list, fully unrolled (register arrays)
#define INSERT10(TV, TI, V, ID)                                         \
  do {                                                                  \
    _Pragma("unroll")                                                   \
    for (int j_ = 9; j_ >= 1; --j_) {                                   \
      bool bj_ = BETTER(V, ID, TV[j_], TI[j_]);                         \
      if (bj_) {                                                        \
        bool bp_ = BETTER(V, ID, TV[j_-1], TI[j_-1]);                   \
        TV[j_] = bp_ ? TV[j_-1] : (V);                                  \
        TI[j_] = bp_ ? TI[j_-1] : (ID);                                 \
      }                                                                 \
    }                                                                   \
    if (BETTER(V, ID, TV[0], TI[0])) { TV[0] = (V); TI[0] = (ID); }     \
  } while (0)

// ---------------- phase 0a: normalize x -> bf16 ----------------
__global__ __launch_bounds__(256) void knn_norm_x(const float* __restrict__ x,
                                                  uint32_t* __restrict__ xnbf) {
  const int wave = threadIdx.x >> 5, lane = threadIdx.x & 31;
  const int row  = blockIdx.x * 8 + wave;            // 256 blocks * 8 = 2048
  const float4* p = (const float4*)(x + (size_t)row * DIM) + lane * 4;
  float4 a = p[0], b = p[1], c = p[2], d = p[3];
  float ss = a.x*a.x + a.y*a.y + a.z*a.z + a.w*a.w
           + b.x*b.x + b.y*b.y + b.z*b.z + b.w*b.w
           + c.x*c.x + c.y*c.y + c.z*c.z + c.w*c.w
           + d.x*d.x + d.y*d.y + d.z*d.z + d.w*d.w;
  #pragma unroll
  for (int m = 16; m >= 1; m >>= 1) ss += __shfl_xor(ss, m, 32);
  float rn = 1.0f / fmaxf(sqrtf(ss), 1e-8f);
  uint32_t o0 = f2bf_bits(a.x*rn) | (f2bf_bits(a.y*rn) << 16);
  uint32_t o1 = f2bf_bits(a.z*rn) | (f2bf_bits(a.w*rn) << 16);
  uint32_t o2 = f2bf_bits(b.x*rn) | (f2bf_bits(b.y*rn) << 16);
  uint32_t o3 = f2bf_bits(b.z*rn) | (f2bf_bits(b.w*rn) << 16);
  uint32_t o4 = f2bf_bits(c.x*rn) | (f2bf_bits(c.y*rn) << 16);
  uint32_t o5 = f2bf_bits(c.z*rn) | (f2bf_bits(c.w*rn) << 16);
  uint32_t o6 = f2bf_bits(d.x*rn) | (f2bf_bits(d.y*rn) << 16);
  uint32_t o7 = f2bf_bits(d.z*rn) | (f2bf_bits(d.w*rn) << 16);
  uint4* dst = (uint4*)xnbf + (size_t)row * 64 + lane * 2; // 64 uint4 per row
  dst[0] = make_uint4(o0, o1, o2, o3);
  dst[1] = make_uint4(o4, o5, o6, o7);
}

// ---------------- phase 0b: embedding reciprocal norms ----------------
__global__ __launch_bounds__(256) void knn_norm_e(const float* __restrict__ emb,
                                                  float* __restrict__ rnormE) {
  const int wave = threadIdx.x >> 5, lane = threadIdx.x & 31;
  const int row  = blockIdx.x * 8 + wave;            // 12500 blocks * 8 = 100000
  const float4* p = (const float4*)(emb + (size_t)row * DIM) + lane * 4;
  float4 a = p[0], b = p[1], c = p[2], d = p[3];
  float ss = a.x*a.x + a.y*a.y + a.z*a.z + a.w*a.w
           + b.x*b.x + b.y*b.y + b.z*b.z + b.w*b.w
           + c.x*c.x + c.y*c.y + c.z*c.z + c.w*c.w
           + d.x*d.x + d.y*d.y + d.z*d.z + d.w*d.w;
  #pragma unroll
  for (int m = 16; m >= 1; m >>= 1) ss += __shfl_xor(ss, m, 32);
  if (lane == 0) rnormE[row] = 1.0f / fmaxf(sqrtf(ss), 1e-8f);
}

// ---------------- phase 1: WMMA GEMM + per-chunk top-10 ----------------
// grid (CHUNKS, MTILES), 256 threads (8 waves -> 2 waves/SIMD). One shared
// 128x512 bf16 A tile in LDS (staged by the Tensor Data Mover when available);
// each wave computes 128x16 sim tiles with v_wmma_f32_16x16x32_bf16, keeps
// per-query top-10 in registers, then the workgroup merges 8 wave lists down
// to 10 candidates per query in LDS before writing to global.
__global__ __launch_bounds__(256) void knn_sim_topk(const uint32_t* __restrict__ xnbf,
                                                    const float* __restrict__ emb,
                                                    const float* __restrict__ rnormE,
                                                    float* __restrict__ candV,
                                                    int* __restrict__ candI) {
  extern __shared__ char smem[];
  const int chunk = blockIdx.x;
  const int mbase = blockIdx.y * MB;
  const int tid  = threadIdx.x;
  const int lane = tid & 31, wave = tid >> 5;
  const int h = lane >> 4, n = lane & 15;

  // ---- stage A tile (128 x 512 bf16) into LDS, padded row stride 520 bf16 ----
#if USE_TDM
  if (tid < 32) {
    // Tensor DMA descriptor (ISA 08 sec 8.3/8.4): 2-D tile, data_size=2B,
    // tensor/tile = 512 x 128, dim0 stride = 512 elems. LDS padding:
    // pad_enable=1, pad_interval=7 (256 DWORDs = one 1024B row),
    // pad_amount=3 (4 DWORDs = 16B) -> LDS row stride 1040B = 520 bf16.
    uint64_t ga = (uint64_t)(xnbf) + (uint64_t)mbase * (DIM * 2);
    u32x4 g0;
    g0[0] = 1u;                                   // count=1 (valid), user mode
    g0[1] = 0u;                                   // lds_addr = dynamic LDS base
    g0[2] = (unsigned int)(ga & 0xFFFFFFFFu);     // global_addr[31:0]
    g0[3] = (unsigned int)((ga >> 32) & 0x1FFFFFFu) | (2u << 30); // addr[56:32], type=2
    i32x8 g1;
    g1[0] = (int)((1u << 16) | (1u << 20) | (7u << 22) | (3u << 25));
    g1[1] = (int)(512u << 16);                    // tensor_dim0[15:0]=512 @ [63:48]
    g1[2] = (int)(128u << 16);                    // tensor_dim1[15:0]=128 @ [95:80]
    g1[3] = (int)(512u << 16);                    // tile_dim0=512 @ [127:112]
    g1[4] = (int)(128u);                          // tile_dim1=128 @ [143:128]
    g1[5] = (int)(512u);                          // tensor_dim0_stride[31:0]=512
    g1[6] = 0; g1[7] = 0;
    i32x4 z4 = {0, 0, 0, 0};
    i32x8 z8 = {0, 0, 0, 0, 0, 0, 0, 0};
    // clang-23 / therock form: (g0, g1, g2, g3, extra, cpol)
    __builtin_amdgcn_tensor_load_to_lds(g0, g1, z4, z4, z8, 0);
    __builtin_amdgcn_s_wait_tensorcnt(0);
  }
#else
  {
    uint4* ldsA4 = (uint4*)smem;                   // 65 uint4 per row
    const uint4* gA = (const uint4*)xnbf;          // 64 uint4 per row
    for (int i = tid; i < MB * 64; i += 256) {
      int r = i >> 6, c = i & 63;
      ldsA4[r * 65 + c] = gA[(size_t)(mbase + r) * 64 + c];
    }
  }
#endif
  __syncthreads();

  const __bf16* ldsAbf = (const __bf16*)smem;
  float* simW = (float*)(smem + LDSA_BYTES) + wave * (MB * SIM_STRIDE);

  // per-lane top-10 for 4 owned queries (rows lane*4 + s)
  float tv[4][10]; int ti[4][10];
  #pragma unroll
  for (int s = 0; s < 4; ++s)
    #pragma unroll
    for (int j = 0; j < 10; ++j) { tv[s][j] = -3.4e38f; ti[s][j] = 0x7FFFFFFF; }

  for (int t = wave; t < TILESPC; t += WAVES) {
    const int nt = chunk * NPC + t * 16;             // base embedding row of tile
    v8f acc[8];
    {
      v8f z = {};
      #pragma unroll
      for (int m = 0; m < 8; ++m) acc[m] = z;
    }
    const bool doPf = (t + WAVES) < TILESPC;
    const float* pfBase = emb + (size_t)(nt + 16 * WAVES + n) * DIM + 16 * h;

    for (int kb = 0; kb < DIM; kb += 32) {
      // prefetch this lane's 64B segment of the NEXT tile this wave will do
      if (doPf) __builtin_prefetch(pfBase + kb, 0, 3);

      // B fragment: lane reads emb[nt+n][kb+16h .. +15] (16 f32) -> bf16 RNE
      // via native casts (v_cvt_pk_bf16_f32).
      // B layout used here, same as A (32x16 bf16): elem e <-> K = kb + 16h + e, N = n.
      const float4* bp = (const float4*)(emb + (size_t)(nt + n) * DIM + kb + 16 * h);
      float4 b0 = bp[0], b1 = bp[1], b2 = bp[2], b3 = bp[3];
      BF16x16 bf;
      bf.v[0]  = (__bf16)b0.x; bf.v[1]  = (__bf16)b0.y;
      bf.v[2]  = (__bf16)b0.z; bf.v[3]  = (__bf16)b0.w;
      bf.v[4]  = (__bf16)b1.x; bf.v[5]  = (__bf16)b1.y;
      bf.v[6]  = (__bf16)b1.z; bf.v[7]  = (__bf16)b1.w;
      bf.v[8]  = (__bf16)b2.x; bf.v[9]  = (__bf16)b2.y;
      bf.v[10] = (__bf16)b2.z; bf.v[11] = (__bf16)b2.w;
      bf.v[12] = (__bf16)b3.x; bf.v[13] = (__bf16)b3.y;
      bf.v[14] = (__bf16)b3.z; bf.v[15] = (__bf16)b3.w;

      // Preload ALL 8 A fragments into distinct registers so the 16 ds_load_b128
      // issue back-to-back and the 8 WMMAs use graduated s_wait_dscnt.
      // A layout (16x32 bf16): elem e <-> K = kb+16h+e; M = n.
      BF16x16 af[8];
      #pragma unroll
      for (int m = 0; m < 8; ++m) {
        const uint4* ap =
            (const uint4*)(ldsAbf + (size_t)(m * 16 + n) * LDSA_STRIDE + kb + 16 * h);
        af[m].q[0] = ap[0];   // K = kb+16h   .. +7
        af[m].q[1] = ap[1];   // K = kb+16h+8 .. +7   (+16 bytes)
      }
      #pragma unroll
      for (int m = 0; m < 8; ++m)
        acc[m] = __builtin_amdgcn_wmma_f32_16x16x32_bf16(
            false, af[m].v, false, bf.v, (short)0, acc[m], false, false);
    }

    // scale by 1/||e_n|| and stage 128x16 sim tile to this wave's LDS region.
    // D layout: acc[m] elem r -> (row = m*16 + r + 8h, col = n)
    const float rn = rnormE[nt + n];
    #pragma unroll
    for (int m = 0; m < 8; ++m)
      #pragma unroll
      for (int r = 0; r < 8; ++r)
        simW[(m * 16 + r + 8 * h) * SIM_STRIDE + n] = acc[m][r] * rn;
    // wave-private LDS region: compiler inserts s_wait_dscnt for the RAW below.

    #pragma unroll
    for (int s = 0; s < 4; ++s) {
      const float* sp = simW + (lane * 4 + s) * SIM_STRIDE;
      for (int nn = 0; nn < 16; ++nn) {
        float v = sp[nn];
        int id = nt + nn;
        if (BETTER(v, id, tv[s][9], ti[s][9])) {
          INSERT10(tv[s], ti[s], v, id);
        }
      }
    }
  }

  // ---- merge the 8 waves' lists to 10 candidates per query (reuse A LDS) ----
  __syncthreads();                                   // all waves done reading A
  float* mergeV = (float*)smem;                      // [128][MERGE_STRIDE]
  int*   mergeI = (int*)(smem + MB * MERGE_STRIDE * 4);
  #pragma unroll
  for (int s = 0; s < 4; ++s) {
    const int row = lane * 4 + s;
    #pragma unroll
    for (int j = 0; j < 10; ++j) {
      mergeV[row * MERGE_STRIDE + wave * 10 + j] = tv[s][j];
      mergeI[row * MERGE_STRIDE + wave * 10 + j] = ti[s][j];
    }
  }
  __syncthreads();

  if (tid < MB) {
    const int row = tid;
    float fv[10]; int fi[10];
    #pragma unroll
    for (int j = 0; j < 10; ++j) { fv[j] = -3.4e38f; fi[j] = 0x7FFFFFFF; }
    for (int c = 0; c < WAVES * 10; ++c) {
      float v = mergeV[row * MERGE_STRIDE + c];
      int  id = mergeI[row * MERGE_STRIDE + c];
      if (BETTER(v, id, fv[9], fi[9])) INSERT10(fv, fi, v, id);
    }
    const size_t base = ((size_t)(mbase + row) * CHUNKS + chunk) * 10;
    #pragma unroll
    for (int j = 0; j < 10; ++j) { candV[base + j] = fv[j]; candI[base + j] = fi[j]; }
  }
}

// ---------------- phase 2: merge candidates, mode of labels ----------------
__global__ __launch_bounds__(256) void knn_finalize(const float* __restrict__ candV,
                                                    const int* __restrict__ candI,
                                                    const int* __restrict__ labels,
                                                    float* __restrict__ out) {
  const int q = blockIdx.x * blockDim.x + threadIdx.x;
  if (q >= NQ) return;
  float tv[10]; int ti[10];
  #pragma unroll
  for (int j = 0; j < 10; ++j) { tv[j] = -3.4e38f; ti[j] = 0x7FFFFFFF; }
  const float* cv = candV + (size_t)q * CANDQ;
  const int*   ci = candI + (size_t)q * CANDQ;
  for (int c = 0; c < CANDQ; ++c) {
    float v = cv[c]; int id = ci[c];
    if (BETTER(v, id, tv[9], ti[9])) INSERT10(tv, ti, v, id);
  }
  int lab[10];
  #pragma unroll
  for (int k = 0; k < 10; ++k) lab[k] = labels[ti[k]];
  // torch.mode semantics: most frequent, ties -> smallest label value
  int bestC = -1, bestL = 0x7FFFFFFF;
  #pragma unroll
  for (int k = 0; k < 10; ++k) {
    int cnt = 0;
    #pragma unroll
    for (int j = 0; j < 10; ++j) cnt += (lab[j] == lab[k]);
    if (cnt > bestC || (cnt == bestC && lab[k] < bestL)) { bestC = cnt; bestL = lab[k]; }
  }
  out[q] = (float)bestL;
}

// ---------------- host launcher ----------------
extern "C" void kernel_launch(void* const* d_in, const int* in_sizes, int n_in,
                              void* d_out, int out_size, void* d_ws, size_t ws_size,
                              hipStream_t stream) {
  (void)in_sizes; (void)n_in; (void)out_size; (void)ws_size;
  const float* x      = (const float*)d_in[0];
  const float* emb    = (const float*)d_in[1];
  const int*   labels = (const int*)d_in[2];
  float*       out    = (float*)d_out;

  // workspace layout (~6.6 MB total)
  char* ws = (char*)d_ws;
  uint32_t* xnbf   = (uint32_t*)ws;                              // 2,097,152 B
  float*    rnormE = (float*)(ws + 2097152);                     //   400,000 B (pad to 401,408)
  float*    candV  = (float*)(ws + 2097152 + 401408);            // 2,048,000 B
  int*      candI  = (int*)  (ws + 2097152 + 401408 + 2048000);  // 2,048,000 B

  knn_norm_x  <<<dim3(NQ / 8),         dim3(256), 0,         stream>>>(x, xnbf);
  knn_norm_e  <<<dim3(NE / 8),         dim3(256), 0,         stream>>>(emb, rnormE);
  knn_sim_topk<<<dim3(CHUNKS, MTILES), dim3(256), LDS_BYTES, stream>>>(xnbf, emb, rnormE,
                                                                       candV, candI);
  knn_finalize<<<dim3(NQ / 256),       dim3(256), 0,         stream>>>(candV, candI, labels, out);
}
